// PM_Attention_Expert_53824530153847
// MI455X (gfx1250) — matrix operand order repaired and checked by reading, not compiled, not run on hardware
//
#include <hip/hip_runtime.h>
#include <stdint.h>

// ---------------- problem dims ----------------
#define SEQ    128
#define BATCH  256
#define EDIM   512
#define FDIM   2048
#define HEADS  8
#define HDIM   64
#define TOK    (SEQ * BATCH)          // 32768 tokens
#define MAXN   0.99999f               // 1 - 1e-5 Poincare clip

// ---------------- WMMA types (CDNA5 gfx1250, wave32) ----------------
typedef __bf16 v16bf __attribute__((ext_vector_type(16)));
typedef float  v8f   __attribute__((ext_vector_type(8)));

union Frag { v16bf v; unsigned short u[16]; };
union Acc  { v8f v; float f[8]; };

// ---------------- bf16 helpers (RNE) ----------------
__device__ __forceinline__ unsigned short f2bf(float f) {
  union { float f; unsigned u; } x; x.f = f;
  unsigned r = x.u + 0x7FFFu + ((x.u >> 16) & 1u);
  return (unsigned short)(r >> 16);
}

// ---------------- CDNA5 async global->LDS copy (ASYNCcnt path) ----------
// dsaddr = low 32 bits of flat LDS pointer
__device__ __forceinline__ void async_load_b128(void* lds_ptr, const void* gaddr) {
  unsigned ldsoff = (unsigned)(uintptr_t)lds_ptr;
  asm volatile("global_load_async_to_lds_b128 %0, %1, off"
               :: "v"(ldsoff), "v"(gaddr) : "memory");
}
__device__ __forceinline__ void wait_async0() {
  asm volatile("s_wait_asynccnt 0" ::: "memory");
}

// ---------------- block reduction (256 threads) ----------------
__device__ __forceinline__ float blk_sum(float v, float* red) {
  int t = threadIdx.x;
  red[t] = v;
  __syncthreads();
#pragma unroll
  for (int s = 128; s > 0; s >>= 1) {
    if (t < s) red[t] += red[t + s];
    __syncthreads();
  }
  float r = red[0];
  __syncthreads();
  return r;
}

// =====================================================================
// f32 -> bf16 weight conversion
// =====================================================================
__global__ __launch_bounds__(256) void k_cvt(const float* __restrict__ in,
                                             unsigned short* __restrict__ outp, int n) {
  int i = blockIdx.x * 256 + threadIdx.x;
  if (i < n) outp[i] = f2bf(in[i]);
}

// =====================================================================
// pre kernel: h = expmap0(layernorm(logmap0(x)));  out bf16 h, ||h||
// one block per token, NP = D/256 elements per thread
// =====================================================================
template <int NP>
__global__ __launch_bounds__(256) void k_pre_ln(const float* __restrict__ X,
                                                const float* __restrict__ g,
                                                const float* __restrict__ beta,
                                                unsigned short* __restrict__ Hb,
                                                float* __restrict__ Hn) {
  __shared__ float red[256];
  const int D = NP * 256;
  int tok = blockIdx.x, t = threadIdx.x;
  const float* x = X + (size_t)tok * D;
  float xv[NP];
  float s2 = 0.f;
#pragma unroll
  for (int i = 0; i < NP; ++i) { float v = x[i * 256 + t]; xv[i] = v; s2 += v * v; }
  float n2 = blk_sum(s2, red) + 1e-30f;
  float n  = sqrtf(n2);
  float ls = atanhf(fminf(n, MAXN)) / n;          // logmap0 scale
  float su = 0.f, sq = 0.f;
#pragma unroll
  for (int i = 0; i < NP; ++i) { xv[i] *= ls; su += xv[i]; sq += xv[i] * xv[i]; }
  su = blk_sum(su, red);
  sq = blk_sum(sq, red);
  float mean = su / D;
  float var  = sq / D - mean * mean;
  float inv  = rsqrtf(var + 1e-5f);
  float st = 0.f;
#pragma unroll
  for (int i = 0; i < NP; ++i) {
    int idx = i * 256 + t;
    float tv = (xv[i] - mean) * inv * g[idx] + beta[idx];
    xv[i] = tv; st += tv * tv;
  }
  st = blk_sum(st, red);
  float tn = sqrtf(st + 1e-30f);
  float es = tanhf(tn) / tn;                      // expmap0 scale
#pragma unroll
  for (int i = 0; i < NP; ++i) Hb[(size_t)tok * D + i * 256 + t] = f2bf(es * xv[i]);
  if (t == 0) Hn[tok] = tanhf(tn) * sqrtf(st) / tn;
}

// =====================================================================
// mobius linear core: z = projx(mobius_add(mobius_matvec_scale(mx), b))
// =====================================================================
template <int NP>
__device__ __forceinline__ float mobius_linear_z(const float* __restrict__ mx_row,
                                                 const float* __restrict__ bias,
                                                 float xn, float* red, float zv[NP]) {
  int t = threadIdx.x;
  float bv[NP];
  float smm = 0.f;
#pragma unroll
  for (int i = 0; i < NP; ++i) {
    int idx = i * 256 + t;
    float m = mx_row[idx]; zv[i] = m; bv[i] = bias[idx];
    smm += m * m;
  }
  smm = blk_sum(smm, red);
  float mxn = sqrtf(smm + 1e-30f);
  float ar  = atanhf(fminf(xn, MAXN));
  float r   = tanhf(mxn / fmaxf(xn, 1e-30f) * ar) / mxn;   // mobius_matvec scale
  float sxy = 0.f, sbb = 0.f;
#pragma unroll
  for (int i = 0; i < NP; ++i) { zv[i] *= r; sxy += zv[i] * bv[i]; sbb += bv[i] * bv[i]; }
  sxy = blk_sum(sxy, red);
  sbb = blk_sum(sbb, red);
  float y2   = r * r * smm;
  float cnum = 1.f + 2.f * sxy + sbb;
  float cy   = 1.f - y2;
  float den  = fmaxf(1.f + 2.f * sxy + y2 * sbb, 1e-15f);
  float szz = 0.f;
#pragma unroll
  for (int i = 0; i < NP; ++i) { float z = (cnum * zv[i] + cy * bv[i]) / den; zv[i] = z; szz += z * z; }
  szz = blk_sum(szz, red);
  float zn = sqrtf(szz + 1e-30f);
  if (zn > MAXN) {
    float s = MAXN / zn;
#pragma unroll
    for (int i = 0; i < NP; ++i) zv[i] *= s;
    zn = MAXN;
  }
  return zn;
}

// epilogue: q/k/v -> logmap0(man_linear(...)) as bf16 tangent vectors
template <int NP>
__global__ __launch_bounds__(256) void k_mob_tan(const float* __restrict__ MX,
                                                 const float* __restrict__ Xn,
                                                 const float* __restrict__ bias,
                                                 unsigned short* __restrict__ Out) {
  __shared__ float red[256];
  const int D = NP * 256;
  int tok = blockIdx.x, t = threadIdx.x;
  float zv[NP];
  float zn = mobius_linear_z<NP>(MX + (size_t)tok * D, bias, Xn[tok], red, zv);
  float ls = atanhf(fminf(zn, MAXN)) / zn;
#pragma unroll
  for (int i = 0; i < NP; ++i) Out[(size_t)tok * D + i * 256 + t] = f2bf(ls * zv[i]);
}

// epilogue: manifold out + mobius residual add + projx  (attention Wo path)
template <int NP>
__global__ __launch_bounds__(256) void k_mob_manifold_res(const float* __restrict__ MX,
                                                          const float* __restrict__ Xn,
                                                          const float* __restrict__ bias,
                                                          const float* __restrict__ Res,
                                                          float* __restrict__ Out) {
  __shared__ float red[256];
  const int D = NP * 256;
  int tok = blockIdx.x, t = threadIdx.x;
  float zv[NP];
  float zn = mobius_linear_z<NP>(MX + (size_t)tok * D, bias, Xn[tok], red, zv);
  float z2 = zn * zn;
  const float* rrow = Res + (size_t)tok * D;
  float rv[NP];
  float sxr = 0.f, srr = 0.f;
#pragma unroll
  for (int i = 0; i < NP; ++i) { rv[i] = rrow[i * 256 + t]; sxr += zv[i] * rv[i]; srr += rv[i] * rv[i]; }
  sxr = blk_sum(sxr, red);
  srr = blk_sum(srr, red);
  float cnum = 1.f + 2.f * sxr + srr;
  float cz   = 1.f - z2;
  float den  = fmaxf(1.f + 2.f * sxr + z2 * srr, 1e-15f);
  float sww = 0.f;
#pragma unroll
  for (int i = 0; i < NP; ++i) { float w = (cnum * zv[i] + cz * rv[i]) / den; zv[i] = w; sww += w * w; }
  sww = blk_sum(sww, red);
  float wn = sqrtf(sww + 1e-30f);
  float sc = (wn > MAXN) ? (MAXN / wn) : 1.f;
#pragma unroll
  for (int i = 0; i < NP; ++i) Out[(size_t)tok * D + i * 256 + t] = sc * zv[i];
}

// epilogue fc1: mob_relu(man_linear(...)) -> bf16 + new norm (Xn overwritten)
template <int NP>
__global__ __launch_bounds__(256) void k_mob_act(const float* __restrict__ MX,
                                                 float* __restrict__ Xn,
                                                 const float* __restrict__ bias,
                                                 unsigned short* __restrict__ Out) {
  __shared__ float red[256];
  const int D = NP * 256;
  int tok = blockIdx.x, t = threadIdx.x;
  float xn = Xn[tok];               // read before any write
  float zv[NP];
  float zn = mobius_linear_z<NP>(MX + (size_t)tok * D, bias, xn, red, zv);
  float ls = atanhf(fminf(zn, MAXN)) / zn;   // logmap0
  float st = 0.f;
#pragma unroll
  for (int i = 0; i < NP; ++i) { float tv = fmaxf(ls * zv[i], 0.f); zv[i] = tv; st += tv * tv; }
  st = blk_sum(st, red);
  float tn = sqrtf(st + 1e-30f);
  float es = tanhf(tn) / tn;                 // expmap0 (projx identity: tanh<1)
#pragma unroll
  for (int i = 0; i < NP; ++i) Out[(size_t)tok * D + i * 256 + t] = f2bf(es * zv[i]);
  if (t == 0) Xn[tok] = tanhf(tn) * sqrtf(st) / tn;
}

// epilogue fc2: mob_relu(man_linear(...)) then mobius residual + projx -> f32 out
template <int NP>
__global__ __launch_bounds__(256) void k_mob_final(const float* __restrict__ MX,
                                                   const float* __restrict__ Xn,
                                                   const float* __restrict__ bias,
                                                   const float* __restrict__ Res,
                                                   float* __restrict__ Out) {
  __shared__ float red[256];
  const int D = NP * 256;
  int tok = blockIdx.x, t = threadIdx.x;
  float zv[NP];
  float zn = mobius_linear_z<NP>(MX + (size_t)tok * D, bias, Xn[tok], red, zv);
  float ls = atanhf(fminf(zn, MAXN)) / zn;
  float st = 0.f;
#pragma unroll
  for (int i = 0; i < NP; ++i) { float tv = fmaxf(ls * zv[i], 0.f); zv[i] = tv; st += tv * tv; }
  st = blk_sum(st, red);
  float tn = sqrtf(st + 1e-30f);
  float es = tanhf(tn) / tn;
  float hn = tanhf(tn) * sqrtf(st) / tn;
  float h2 = hn * hn;
  const float* rrow = Res + (size_t)tok * D;
  float rv[NP];
  float sxr = 0.f, srr = 0.f;
#pragma unroll
  for (int i = 0; i < NP; ++i) {
    rv[i] = rrow[i * 256 + t];
    float h = es * zv[i]; zv[i] = h;
    sxr += h * rv[i]; srr += rv[i] * rv[i];
  }
  sxr = blk_sum(sxr, red);
  srr = blk_sum(srr, red);
  float cnum = 1.f + 2.f * sxr + srr;
  float ch   = 1.f - h2;
  float den  = fmaxf(1.f + 2.f * sxr + h2 * srr, 1e-15f);
  float sww = 0.f;
#pragma unroll
  for (int i = 0; i < NP; ++i) { float w = (cnum * zv[i] + ch * rv[i]) / den; zv[i] = w; sww += w * w; }
  sww = blk_sum(sww, red);
  float wn = sqrtf(sww + 1e-30f);
  float sc = (wn > MAXN) ? (MAXN / wn) : 1.f;
#pragma unroll
  for (int i = 0; i < NP; ++i) Out[(size_t)tok * D + i * 256 + t] = sc * zv[i];
}

// expmap0 of attention output -> bf16 + norm (GEMM input for Wo)
template <int NP>
__global__ __launch_bounds__(256) void k_expmap_norm(const float* __restrict__ O,
                                                     unsigned short* __restrict__ Hb,
                                                     float* __restrict__ Hn) {
  __shared__ float red[256];
  const int D = NP * 256;
  int tok = blockIdx.x, t = threadIdx.x;
  const float* o = O + (size_t)tok * D;
  float ov[NP];
  float st = 0.f;
#pragma unroll
  for (int i = 0; i < NP; ++i) { float v = o[i * 256 + t]; ov[i] = v; st += v * v; }
  st = blk_sum(st, red);
  float tn = sqrtf(st + 1e-30f);
  float es = tanhf(tn) / tn;
#pragma unroll
  for (int i = 0; i < NP; ++i) Hb[(size_t)tok * D + i * 256 + t] = f2bf(es * ov[i]);
  if (t == 0) Hn[tok] = tanhf(tn) * sqrtf(st) / tn;
}

// =====================================================================
// bf16 WMMA GEMM: C[M,N] = A[M,K] * W[N,K]^T   (x @ W.T, both K-major)
// block = 256 threads (8 waves); tile 256(M) x 64(N); K-step 32.
// Double-buffered async global->LDS staging (ASYNCcnt), wave w owns rows
// [w*32, w*32+32) as two 16-row WMMA tiles x 4 N-tiles -> 8 WMMA / K-step.
// =====================================================================
__global__ __launch_bounds__(256) void k_gemm_bf16(const unsigned short* __restrict__ A,
                                                   const unsigned short* __restrict__ W,
                                                   float* __restrict__ C,
                                                   int M, int N, int K) {
  __shared__ __align__(16) unsigned short As[2][256 * 32];  // 2 x 16KB
  __shared__ __align__(16) unsigned short Bs[2][64 * 32];   // 2 x 4KB
  int tid = threadIdx.x;
  int row0 = blockIdx.y * 256, col0 = blockIdx.x * 64;
  int wid = tid >> 5, lane = tid & 31;
  int lm = lane & 15, lh = lane >> 4;

  Acc acc[2][4];
#pragma unroll
  for (int mt = 0; mt < 2; ++mt)
#pragma unroll
    for (int nt = 0; nt < 4; ++nt)
#pragma unroll
      for (int i = 0; i < 8; ++i) acc[mt][nt].f[i] = 0.f;

  // staging map: A row = tid (256 rows, 64B each = 4x b128); B row = tid>>2, chunk tid&3
  const unsigned short* arow = A + (size_t)(row0 + tid) * K;
  const unsigned short* brow = W + (size_t)(col0 + (tid >> 2)) * K + (tid & 3) * 8;
  int aofs = tid * 32, bofs = (tid >> 2) * 32 + (tid & 3) * 8;

  // prologue: async-stage tile 0 into buffer 0
#pragma unroll
  for (int j = 0; j < 4; ++j) async_load_b128(&As[0][aofs + j * 8], arow + j * 8);
  async_load_b128(&Bs[0][bofs], brow);
  wait_async0();
  __syncthreads();

  int buf = 0;
  for (int kk = 0; kk < K; kk += 32) {
    // async-stage next K tile into the other buffer while we compute
    if (kk + 32 < K) {
#pragma unroll
      for (int j = 0; j < 4; ++j)
        async_load_b128(&As[buf ^ 1][aofs + j * 8], arow + kk + 32 + j * 8);
      async_load_b128(&Bs[buf ^ 1][bofs], brow + kk + 32);
      if (kk + 64 < K) __builtin_prefetch(arow + kk + 64, 0, 1);  // global_prefetch_b8
    }
    // A fragments: 16x32 bf16, contiguous k order shared with B (lane -> row lm, K-half by lane>>4)
    int kb = lh * 8, kb2 = lh * 16;
    Frag a[2];
#pragma unroll
    for (int mt = 0; mt < 2; ++mt)
#pragma unroll
      for (int j = 0; j < 8; ++j) {
        a[mt].u[j]     = As[buf][(wid * 32 + mt * 16 + lm) * 32 + kb2 + j];
        a[mt].u[j + 8] = As[buf][(wid * 32 + mt * 16 + lm) * 32 + kb2 + 8 + j];
      }
#pragma unroll
    for (int nt = 0; nt < 4; ++nt) {
      Frag b;
#pragma unroll
      for (int j = 0; j < 16; ++j) b.u[j] = Bs[buf][(nt * 16 + lm) * 32 + kb2 + j];
#pragma unroll
      for (int mt = 0; mt < 2; ++mt)
        acc[mt][nt].v = __builtin_amdgcn_wmma_f32_16x16x32_bf16(
            false, a[mt].v, false, b.v, (short)0, acc[mt][nt].v, false, false);
    }
    wait_async0();      // our next-tile async copies landed in LDS
    __syncthreads();    // everyone's landed; current buffer free for reuse
    buf ^= 1;
  }
  // C/D layout: VGPR i -> M = i + 8*lh, N = lm
#pragma unroll
  for (int mt = 0; mt < 2; ++mt)
#pragma unroll
    for (int nt = 0; nt < 4; ++nt)
#pragma unroll
      for (int i = 0; i < 8; ++i) {
        int r = row0 + wid * 32 + mt * 16 + i + lh * 8;
        int c = col0 + nt * 16 + lm;
        C[(size_t)r * N + c] = acc[mt][nt].f[i];
      }
}

// =====================================================================
// tan-space attention, one workgroup per (b,h). S=T=128, hd=64.
// dynamic LDS 112KB: [Qs 16K | Ks 16K] overlaid by Ss 64K, Ps 32K, Vs 16K
// =====================================================================
__global__ __launch_bounds__(256) void k_attn(const unsigned short* __restrict__ Q,
                                              const unsigned short* __restrict__ Kt,
                                              const unsigned short* __restrict__ V,
                                              float* __restrict__ O) {
  extern __shared__ char smem[];
  unsigned short* Qs = (unsigned short*)smem;                   // [128][64]
  unsigned short* Ks = Qs + 128 * 64;                           // [128][64]
  float*          Ss = (float*)smem;                            // [128][128] overlays Qs/Ks
  unsigned short* Ps = (unsigned short*)(smem + 65536);         // [128][128]
  unsigned short* Vs = (unsigned short*)(smem + 65536 + 32768); // [128][64]

  int b = blockIdx.x / HEADS, h = blockIdx.x % HEADS;
  int tid = threadIdx.x;
  int wid = tid >> 5, lane = tid & 31;
  int lm = lane & 15, lh = lane >> 4;

  // stage q,k,v head slices: (s, b, h*64 + d)
  {
    int s = tid >> 1, d0 = (tid & 1) * 32;
    size_t base = (size_t)s * BATCH * EDIM + (size_t)b * EDIM + h * HDIM + d0;
#pragma unroll
    for (int j = 0; j < 4; ++j) {
      *(uint4*)&Qs[s * 64 + d0 + j * 8] = *(const uint4*)(Q + base + j * 8);
      *(uint4*)&Ks[s * 64 + d0 + j * 8] = *(const uint4*)(Kt + base + j * 8);
      *(uint4*)&Vs[s * 64 + d0 + j * 8] = *(const uint4*)(V + base + j * 8);
    }
  }
  __syncthreads();

  // scores: wave w rows [w*16, w*16+16) x 128 cols, K = 64
  Acc sc[8];
#pragma unroll
  for (int nt = 0; nt < 8; ++nt)
#pragma unroll
    for (int i = 0; i < 8; ++i) sc[nt].f[i] = 0.f;

#pragma unroll
  for (int kk = 0; kk < 64; kk += 32) {
    Frag a;
    int kb = lh * 8;
#pragma unroll
    for (int j = 0; j < 8; ++j) {
      a.u[j]     = Qs[(wid * 16 + lm) * 64 + kk + lh * 16 + j];
      a.u[j + 8] = Qs[(wid * 16 + lm) * 64 + kk + lh * 16 + 8 + j];
    }
    int kb2 = lh * 16;
#pragma unroll
    for (int nt = 0; nt < 8; ++nt) {
      Frag bb;
#pragma unroll
      for (int j = 0; j < 16; ++j) bb.u[j] = Ks[(nt * 16 + lm) * 64 + kk + kb2 + j];
      sc[nt].v = __builtin_amdgcn_wmma_f32_16x16x32_bf16(
          false, a.v, false, bb.v, (short)0, sc[nt].v, false, false);
    }
  }
  __syncthreads();   // everyone done reading Qs/Ks before Ss overlays them
#pragma unroll
  for (int nt = 0; nt < 8; ++nt)
#pragma unroll
    for (int i = 0; i < 8; ++i)
      Ss[(wid * 16 + i + lh * 8) * 128 + nt * 16 + lm] = sc[nt].f[i] * 0.125f; // 1/sqrt(64)
  __syncthreads();

  // softmax: threads 0..127 own one row each
  if (tid < 128) {
    float* row = Ss + tid * 128;
    float mx = -3.4e38f;
    for (int c = 0; c < 128; ++c) mx = fmaxf(mx, row[c]);
    float sum = 0.f;
    for (int c = 0; c < 128; ++c) { float e = __expf(row[c] - mx); row[c] = e; sum += e; }
    float inv = 1.f / sum;
    unsigned short* prow = Ps + tid * 128;
    for (int c = 0; c < 128; ++c) prow[c] = f2bf(row[c] * inv);
  }
  __syncthreads();

  // o = P @ V : M=128, K=128, N=64
  Acc oa[4];
#pragma unroll
  for (int nt = 0; nt < 4; ++nt)
#pragma unroll
    for (int i = 0; i < 8; ++i) oa[nt].f[i] = 0.f;

#pragma unroll
  for (int kk = 0; kk < 128; kk += 32) {
    Frag a;
    int kb = lh * 8;
#pragma unroll
    for (int j = 0; j < 8; ++j) {
      a.u[j]     = Ps[(wid * 16 + lm) * 128 + kk + lh * 16 + j];
      a.u[j + 8] = Ps[(wid * 16 + lm) * 128 + kk + lh * 16 + 8 + j];
    }
    int kb2 = lh * 16;
#pragma unroll
    for (int nt = 0; nt < 4; ++nt) {
      Frag bb;
#pragma unroll
      for (int j = 0; j < 16; ++j) bb.u[j] = Vs[(kk + kb2 + j) * 64 + nt * 16 + lm];
      oa[nt].v = __builtin_amdgcn_wmma_f32_16x16x32_bf16(
          false, a.v, false, bb.v, (short)0, oa[nt].v, false, false);
    }
  }
  // store o (f32) back to (s, b, h*64 + d)
#pragma unroll
  for (int nt = 0; nt < 4; ++nt)
#pragma unroll
    for (int i = 0; i < 8; ++i) {
      int s = wid * 16 + i + lh * 8;
      int e = h * HDIM + nt * 16 + lm;
      O[(size_t)s * BATCH * EDIM + (size_t)b * EDIM + e] = oa[nt].f[i];
    }
}

// =====================================================================
// host-side launch
// =====================================================================
extern "C" void kernel_launch(void* const* d_in, const int* in_sizes, int n_in,
                              void* d_out, int out_size, void* d_ws, size_t ws_size,
                              hipStream_t stream) {
  (void)in_sizes; (void)n_in; (void)out_size; (void)ws_size;
  const float* x    = (const float*)d_in[0];
  const float* ln1g = (const float*)d_in[1];
  const float* ln1b = (const float*)d_in[2];
  const float* ln2g = (const float*)d_in[3];
  const float* ln2b = (const float*)d_in[4];
  const float* Wq = (const float*)d_in[5];  const float* bq = (const float*)d_in[6];
  const float* Wk = (const float*)d_in[7];  const float* bk = (const float*)d_in[8];
  const float* Wv = (const float*)d_in[9];  const float* bv = (const float*)d_in[10];
  const float* Wo = (const float*)d_in[11]; const float* bo = (const float*)d_in[12];
  const float* W1 = (const float*)d_in[13]; const float* b1 = (const float*)d_in[14];
  const float* W2 = (const float*)d_in[15]; const float* b2 = (const float*)d_in[16];
  float* out = (float*)d_out;

  char* ws = (char*)d_ws;
  size_t off = 0;
  auto carve = [&](size_t bytes) -> char* {
    char* p = ws + off;
    off += (bytes + 255) & ~(size_t)255;
    return p;
  };
  unsigned short* Wqb = (unsigned short*)carve((size_t)EDIM * EDIM * 2);
  unsigned short* Wkb = (unsigned short*)carve((size_t)EDIM * EDIM * 2);
  unsigned short* Wvb = (unsigned short*)carve((size_t)EDIM * EDIM * 2);
  unsigned short* Wob = (unsigned short*)carve((size_t)EDIM * EDIM * 2);
  unsigned short* W1b = (unsigned short*)carve((size_t)FDIM * EDIM * 2);
  unsigned short* W2b = (unsigned short*)carve((size_t)EDIM * FDIM * 2);
  unsigned short* Abf = (unsigned short*)carve((size_t)TOK * EDIM * 2);
  float*          An  = (float*)carve((size_t)TOK * 4);
  float*          G   = (float*)carve((size_t)TOK * EDIM * 4);
  float*          GF  = (float*)carve((size_t)TOK * FDIM * 4);
  unsigned short* Qb  = (unsigned short*)carve((size_t)TOK * EDIM * 2);
  unsigned short* Kb  = (unsigned short*)carve((size_t)TOK * EDIM * 2);
  unsigned short* Vb  = (unsigned short*)carve((size_t)TOK * EDIM * 2);
  float*          M1  = (float*)carve((size_t)TOK * EDIM * 4);
  unsigned short* AF  = (unsigned short*)carve((size_t)TOK * FDIM * 2);

  dim3 blk(256);
  // weights -> bf16
  int ne = EDIM * EDIM, nf = FDIM * EDIM;
  k_cvt<<<(ne + 255) / 256, blk, 0, stream>>>(Wq, Wqb, ne);
  k_cvt<<<(ne + 255) / 256, blk, 0, stream>>>(Wk, Wkb, ne);
  k_cvt<<<(ne + 255) / 256, blk, 0, stream>>>(Wv, Wvb, ne);
  k_cvt<<<(ne + 255) / 256, blk, 0, stream>>>(Wo, Wob, ne);
  k_cvt<<<(nf + 255) / 256, blk, 0, stream>>>(W1, W1b, nf);
  k_cvt<<<(nf + 255) / 256, blk, 0, stream>>>(W2, W2b, nf);

  // h = expmap0(LN(logmap0(x)))
  k_pre_ln<2><<<TOK, blk, 0, stream>>>(x, ln1g, ln1b, Abf, An);

  dim3 gE(EDIM / 64, TOK / 256);
  // q / k / v projections (GEMM + mobius epilogue -> bf16 tangent)
  k_gemm_bf16<<<gE, blk, 0, stream>>>(Abf, Wqb, G, TOK, EDIM, EDIM);
  k_mob_tan<2><<<TOK, blk, 0, stream>>>(G, An, bq, Qb);
  k_gemm_bf16<<<gE, blk, 0, stream>>>(Abf, Wkb, G, TOK, EDIM, EDIM);
  k_mob_tan<2><<<TOK, blk, 0, stream>>>(G, An, bk, Kb);
  k_gemm_bf16<<<gE, blk, 0, stream>>>(Abf, Wvb, G, TOK, EDIM, EDIM);
  k_mob_tan<2><<<TOK, blk, 0, stream>>>(G, An, bv, Vb);

  // tan-space attention (one WG per (b,h)), 112KB dynamic LDS
  k_attn<<<BATCH * HEADS, blk, 112 * 1024, stream>>>(Qb, Kb, Vb, G);

  // o -> expmap0 -> Wo projection -> manifold + residual x
  k_expmap_norm<2><<<TOK, blk, 0, stream>>>(G, Abf, An);
  k_gemm_bf16<<<gE, blk, 0, stream>>>(Abf, Wob, G, TOK, EDIM, EDIM);
  k_mob_manifold_res<2><<<TOK, blk, 0, stream>>>(G, An, bo, x, M1);

  // FFN: LN2 -> fc1 (E->F) + mob_relu -> fc2 (F->E) + mob_relu + residual
  k_pre_ln<2><<<TOK, blk, 0, stream>>>(M1, ln2g, ln2b, Abf, An);
  dim3 gF(FDIM / 64, TOK / 256);
  k_gemm_bf16<<<gF, blk, 0, stream>>>(Abf, W1b, GF, TOK, FDIM, EDIM);
  k_mob_act<8><<<TOK, blk, 0, stream>>>(GF, An, b1, AF);
  k_gemm_bf16<<<gE, blk, 0, stream>>>(AF, W2b, G, TOK, EDIM, FDIM);
  k_mob_final<2><<<TOK, blk, 0, stream>>>(G, An, b2, M1, out);
}
